// MHA_14310831030696
// MI455X (gfx1250) — matrix operand order repaired and checked by reading, not compiled, not run on hardware
//
#include <hip/hip_runtime.h>

typedef __attribute__((ext_vector_type(16))) _Float16 v16h;
typedef __attribute__((ext_vector_type(8)))  _Float16 v8h;
typedef __attribute__((ext_vector_type(8)))  float    v8f;

#define WMMA_F16(a, b, c) \
  __builtin_amdgcn_wmma_f32_16x16x32_f16(false, (a), false, (b), (short)0, (c), false, false)

static constexpr int S_LEN = 2048;
static constexpr int D_DIM = 1024;
static constexpr int H_NUM = 16;
static constexpr int DK_V  = 64;
static constexpr int B_NUM = 2;

// ---------------------------------------------------------------------------
// f32 -> f16 cast (grid-stride)
// ---------------------------------------------------------------------------
__global__ __launch_bounds__(256) void cast_f32_f16(const float* __restrict__ in,
                                                    _Float16* __restrict__ out, int n) {
  int i = blockIdx.x * 256 + threadIdx.x;
  const int stride = gridDim.x * 256;
  for (; i < n; i += stride) out[i] = (_Float16)in[i];
}

// A-fragment loader: contiguous k order (k = 16*hi + s), matched by the B loads.
// lanes 0-15 hold K {0..15}, lanes 16-31 hold K {16..31} of the same row.
union AFrag { v16h v; v8h h[2]; };
__device__ inline v16h load_a_frag(const _Float16* rowk0, int hi) {
  AFrag f;
  f.h[0] = *(const v8h*)(rowk0 + hi * 16);       // K hi*16 .. hi*16+7
  f.h[1] = *(const v8h*)(rowk0 + hi * 16 + 8);   // K hi*16+8 .. hi*16+15
  return f.v;
}

// ---------------------------------------------------------------------------
// Y[m,n] = sum_k X[m,k] * W[n,k] + bias[n]   (torch Linear: x @ W^T + b)
// X: [4096,1024] f16 row-major; W: [1024,1024] f16 row-major.
// Wave computes a 32x64 tile; block = 8 waves -> 64x256 tile.
// mode 0: Y f16 row-major [m,n]; mode 1: Y f16 transposed per batch: [b, n, s]
// ---------------------------------------------------------------------------
__global__ __launch_bounds__(256) void gemm_xwT_f16out(
    const _Float16* __restrict__ X, const _Float16* __restrict__ W,
    const float* __restrict__ bias, _Float16* __restrict__ Y, int mode) {
  const int lane = threadIdx.x & 31;
  const int wave = threadIdx.x >> 5;
  const int lrow = lane & 15;
  const int hi   = lane >> 4;
  const int m0 = blockIdx.y * 64 + (wave >> 2) * 32;
  const int n0 = blockIdx.x * 256 + (wave & 3) * 64;

  v8f acc[2][4] = {};
  const _Float16* ap0 = X + (size_t)(m0 + lrow) * D_DIM;
  const _Float16* ap1 = ap0 + (size_t)16 * D_DIM;
  const _Float16* bp  = W + (size_t)(n0 + lrow) * D_DIM + hi * 16;

  for (int k = 0; k < D_DIM; k += 32) {
    v16h a0 = load_a_frag(ap0 + k, hi);
    v16h a1 = load_a_frag(ap1 + k, hi);
#pragma unroll
    for (int bn = 0; bn < 4; ++bn) {
      v16h b = *(const v16h*)(bp + (size_t)bn * 16 * D_DIM + k);
      acc[0][bn] = WMMA_F16(a0, b, acc[0][bn]);
      acc[1][bn] = WMMA_F16(a1, b, acc[1][bn]);
    }
  }

#pragma unroll
  for (int am = 0; am < 2; ++am)
#pragma unroll
    for (int bn = 0; bn < 4; ++bn) {
      const int col = n0 + bn * 16 + lrow;
      const float bv = bias[col];
      const int rowbase = m0 + am * 16 + hi * 8;   // C/D layout: row = r + 8*hi
#pragma unroll
      for (int r = 0; r < 8; ++r) {
        const float v = acc[am][bn][r] + bv;
        const int row = rowbase + r;
        if (mode == 0) {
          Y[(size_t)row * D_DIM + col] = (_Float16)v;
        } else {  // V^T: [b, col, s]
          const int bb = row >> 11, ss = row & (S_LEN - 1);
          Y[((size_t)bb * D_DIM + col) * S_LEN + ss] = (_Float16)v;
        }
      }
    }
}

// Same GEMM, f32 output (final projection writes d_out).
__global__ __launch_bounds__(256) void gemm_xwT_f32out(
    const _Float16* __restrict__ X, const _Float16* __restrict__ W,
    const float* __restrict__ bias, float* __restrict__ Y) {
  const int lane = threadIdx.x & 31;
  const int wave = threadIdx.x >> 5;
  const int lrow = lane & 15;
  const int hi   = lane >> 4;
  const int m0 = blockIdx.y * 64 + (wave >> 2) * 32;
  const int n0 = blockIdx.x * 256 + (wave & 3) * 64;

  v8f acc[2][4] = {};
  const _Float16* ap0 = X + (size_t)(m0 + lrow) * D_DIM;
  const _Float16* ap1 = ap0 + (size_t)16 * D_DIM;
  const _Float16* bp  = W + (size_t)(n0 + lrow) * D_DIM + hi * 16;

  for (int k = 0; k < D_DIM; k += 32) {
    v16h a0 = load_a_frag(ap0 + k, hi);
    v16h a1 = load_a_frag(ap1 + k, hi);
#pragma unroll
    for (int bn = 0; bn < 4; ++bn) {
      v16h b = *(const v16h*)(bp + (size_t)bn * 16 * D_DIM + k);
      acc[0][bn] = WMMA_F16(a0, b, acc[0][bn]);
      acc[1][bn] = WMMA_F16(a1, b, acc[1][bn]);
    }
  }

#pragma unroll
  for (int am = 0; am < 2; ++am)
#pragma unroll
    for (int bn = 0; bn < 4; ++bn) {
      const int col = n0 + bn * 16 + lrow;
      const float bv = bias[col];
      const int rowbase = m0 + am * 16 + hi * 8;
#pragma unroll
      for (int r = 0; r < 8; ++r)
        Y[(size_t)(rowbase + r) * D_DIM + col] = acc[am][bn][r] + bv;
    }
}

// ---------------------------------------------------------------------------
// Flash attention (causal), one wave per 16 query rows, 32-key blocks.
// Q,K: f16 [B,S,D] row-major (head h at col h*64); Vt: f16 [B,D,S]; AO: f16 [B,S,D].
// ---------------------------------------------------------------------------
__device__ inline float halfmax16(float v) {
  v = fmaxf(v, __shfl_xor(v, 1, 32));
  v = fmaxf(v, __shfl_xor(v, 2, 32));
  v = fmaxf(v, __shfl_xor(v, 4, 32));
  v = fmaxf(v, __shfl_xor(v, 8, 32));
  return v;
}
__device__ inline float halfsum16(float v) {
  v += __shfl_xor(v, 1, 32);
  v += __shfl_xor(v, 2, 32);
  v += __shfl_xor(v, 4, 32);
  v += __shfl_xor(v, 8, 32);
  return v;
}

__global__ __launch_bounds__(256) void flash_attn_f16(
    const _Float16* __restrict__ Q, const _Float16* __restrict__ Kc,
    const _Float16* __restrict__ Vt, _Float16* __restrict__ AO) {
  __shared__ __align__(32) _Float16 pbuf[8][16 * 32];  // per-wave P staging (C->A relayout)

  const int lane = threadIdx.x & 31;
  const int wave = threadIdx.x >> 5;
  const int lrow = lane & 15;
  const int hi   = lane >> 4;
  const int bh = blockIdx.y;             // 0..31
  const int b = bh >> 4, h = bh & 15;
  const int q0 = blockIdx.x * 128 + wave * 16;

  const _Float16* qb = Q  + ((size_t)b * S_LEN) * D_DIM + (size_t)h * DK_V;
  const _Float16* kb = Kc + ((size_t)b * S_LEN) * D_DIM + (size_t)h * DK_V;
  const _Float16* vb = Vt + ((size_t)b * D_DIM + h * DK_V) * S_LEN;

  const _Float16* qrowp = qb + (size_t)(q0 + lrow) * D_DIM;
  const v16h qa0 = load_a_frag(qrowp, hi);        // d 0..31
  const v16h qa1 = load_a_frag(qrowp + 32, hi);   // d 32..63

  v8f o[4] = {};
  float m_r[8], l_r[8];
#pragma unroll
  for (int r = 0; r < 8; ++r) { m_r[r] = -3.0e38f; l_r[r] = 0.0f; }

  const int nblk = (q0 + 15) / 32 + 1;  // key blocks of 32, causal bound
  for (int j = 0; j < nblk; ++j) {
    const int kbase = j * 32;

    // K^T B-fragments: B[d, kk] = K[kbase+kk, d]; contiguous 32B per lane.
    const _Float16* kpA = kb + (size_t)(kbase + lrow) * D_DIM + hi * 16;
    const _Float16* kpB = kpA + (size_t)16 * D_DIM;
    const v16h kA0 = *(const v16h*)(kpA);
    const v16h kA1 = *(const v16h*)(kpA + 32);
    const v16h kB0 = *(const v16h*)(kpB);
    const v16h kB1 = *(const v16h*)(kpB + 32);

    v8f sA = {}, sB = {};
    sA = WMMA_F16(qa0, kA0, sA);
    sA = WMMA_F16(qa1, kA1, sA);
    sB = WMMA_F16(qa0, kB0, sB);
    sB = WMMA_F16(qa1, kB1, sB);

    const bool need_mask = (kbase + 31) > q0;
    const int colA = kbase + lrow;
    const int colB = colA + 16;
#pragma unroll
    for (int r = 0; r < 8; ++r) {
      const int qrow = q0 + r + hi * 8;
      float a = sA[r] * 0.125f;           // 1/sqrt(64)
      float c = sB[r] * 0.125f;
      if (need_mask) {
        if (colA > qrow) a = -3.0e38f;
        if (colB > qrow) c = -3.0e38f;
      }
      const float mb   = fmaxf(halfmax16(a), halfmax16(c));
      const float mnew = fmaxf(m_r[r], mb);
      const float alpha = __expf(m_r[r] - mnew);
      m_r[r] = mnew;
      const float pA = __expf(a - mnew);
      const float pB = __expf(c - mnew);
      l_r[r] = l_r[r] * alpha + halfsum16(pA) + halfsum16(pB);
#pragma unroll
      for (int n = 0; n < 4; ++n) o[n][r] *= alpha;
      const int prow = r + hi * 8;
      pbuf[wave][prow * 32 + lrow]      = (_Float16)pA;
      pbuf[wave][prow * 32 + 16 + lrow] = (_Float16)pB;
    }

    // P A-fragment from LDS (row-major 16x32), then PV: o[n] += P * V
    const v16h pf = load_a_frag(&pbuf[wave][lrow * 32], hi);
#pragma unroll
    for (int n = 0; n < 4; ++n) {
      const _Float16* vp = vb + (size_t)(n * 16 + lrow) * S_LEN + kbase + hi * 16;
      const v16h vf = *(const v16h*)(vp);
      o[n] = WMMA_F16(pf, vf, o[n]);
    }
  }

#pragma unroll
  for (int r = 0; r < 8; ++r) {
    const float inv = 1.0f / l_r[r];
    const int row = q0 + r + hi * 8;
    _Float16* dst = AO + ((size_t)b * S_LEN + row) * D_DIM + h * DK_V;
#pragma unroll
    for (int n = 0; n < 4; ++n)
      dst[n * 16 + lrow] = (_Float16)(o[n][r] * inv);
  }
}

// ---------------------------------------------------------------------------
// Launcher. Inputs: x, mask, wq, bq, wk, bk, wv, bv, wo, bo (mask handled
// analytically — causal). Workspace: 40 MB (AO reuses the x-f16 buffer).
// ---------------------------------------------------------------------------
extern "C" void kernel_launch(void* const* d_in, const int* in_sizes, int n_in,
                              void* d_out, int out_size, void* d_ws, size_t ws_size,
                              hipStream_t stream) {
  (void)in_sizes; (void)n_in; (void)out_size; (void)ws_size;
  const float* x  = (const float*)d_in[0];
  const float* wq = (const float*)d_in[2];
  const float* bq = (const float*)d_in[3];
  const float* wk = (const float*)d_in[4];
  const float* bk = (const float*)d_in[5];
  const float* wv = (const float*)d_in[6];
  const float* bv = (const float*)d_in[7];
  const float* wo = (const float*)d_in[8];
  const float* bo = (const float*)d_in[9];
  float* out = (float*)d_out;

  char* ws = (char*)d_ws;
  const size_t MB = (size_t)1 << 20;
  _Float16* XH  = (_Float16*)(ws + 0 * MB);    // 8 MB, later reused as AO
  _Float16* WQH = (_Float16*)(ws + 8 * MB);    // 2 MB
  _Float16* WKH = (_Float16*)(ws + 10 * MB);
  _Float16* WVH = (_Float16*)(ws + 12 * MB);
  _Float16* WOH = (_Float16*)(ws + 14 * MB);
  _Float16* QH  = (_Float16*)(ws + 16 * MB);   // 8 MB
  _Float16* KH  = (_Float16*)(ws + 24 * MB);   // 8 MB
  _Float16* VTH = (_Float16*)(ws + 32 * MB);   // 8 MB -> 40 MB total
  _Float16* AOH = XH;

  cast_f32_f16<<<2048, 256, 0, stream>>>(x,  XH,  B_NUM * S_LEN * D_DIM);
  cast_f32_f16<<<512,  256, 0, stream>>>(wq, WQH, D_DIM * D_DIM);
  cast_f32_f16<<<512,  256, 0, stream>>>(wk, WKH, D_DIM * D_DIM);
  cast_f32_f16<<<512,  256, 0, stream>>>(wv, WVH, D_DIM * D_DIM);
  cast_f32_f16<<<512,  256, 0, stream>>>(wo, WOH, D_DIM * D_DIM);

  dim3 gg(D_DIM / 256, (B_NUM * S_LEN) / 64);  // (4, 64)
  gemm_xwT_f16out<<<gg, 256, 0, stream>>>(XH, WQH, bq, QH, 0);
  gemm_xwT_f16out<<<gg, 256, 0, stream>>>(XH, WKH, bk, KH, 0);
  gemm_xwT_f16out<<<gg, 256, 0, stream>>>(XH, WVH, bv, VTH, 1);  // V stored transposed

  flash_attn_f16<<<dim3(S_LEN / 128, B_NUM * H_NUM), 256, 0, stream>>>(QH, KH, VTH, AOH);

  gemm_xwT_f32out<<<gg, 256, 0, stream>>>(AOH, WOH, bo, out);
}
